// Attention_77103252898277
// MI455X (gfx1250) — matrix operand order repaired and checked by reading, not compiled, not run on hardware
//
#include <hip/hip_runtime.h>
#include <cstdint>

// ---------- vector types ----------
typedef __attribute__((ext_vector_type(16))) _Float16 v16h;
typedef __attribute__((ext_vector_type(8)))  _Float16 h8v;
typedef __attribute__((ext_vector_type(8)))  float    v8f;
typedef __attribute__((ext_vector_type(4)))  float    f4v;

#define LDT 40    // LDS stride in halves (80B = 20 banks -> conflict-free for 16 lanes, 16B aligned)
#define LDK 136   // K-tile LDS stride in halves (272B = 4-bank rotation per row, 16B aligned)

static __device__ __forceinline__ v16h combine16(h8v lo, h8v hi) {
  v16h r;
#pragma unroll
  for (int i = 0; i < 8; ++i) { r[i] = lo[i]; r[i + 8] = hi[i]; }
  return r;
}

static __device__ __forceinline__ v8f wmma_f16(v16h a, v16h b, v8f c) {
  // D(f32 16x16) = A(f16 16x32) * B(f16 32x16) + C
  return __builtin_amdgcn_wmma_f32_16x16x32_f16(false, a, false, b, (short)0, c, false, false);
}

// CDNA5 async copy: global -> LDS, 16B per lane, tracked by ASYNCcnt.
// lds_addr is the wave-relative LDS byte address (low 32 bits of the flat
// shared-memory pointer, per the flat->LDS aperture truncation rule).
static __device__ __forceinline__ void async_load_b128(uint32_t lds_addr, const void* gaddr) {
  asm volatile("global_load_async_to_lds_b128 %0, %1, off"
               :
               : "v"(lds_addr), "v"((unsigned long long)(uintptr_t)gaddr)
               : "memory");
}

// =====================================================================
// Generic WMMA GEMM: C[M,Nc] = A[M,Kc] * Bw[Kc,Nc]
// fp32 weights, TA in {float,_Float16} activations, fp32 accumulate,
// TO in {_Float16,float} output. Tiles: BM=128, BN=128, BK=32.
// 256 threads = 8 waves in 2x4; each wave -> 64x32 (4x2 wmma tiles).
// =====================================================================
template <typename TA, typename TO>
__global__ __launch_bounds__(256) void gemm_wmma(
    const TA* __restrict__ A, const float* __restrict__ Bw, TO* __restrict__ C,
    int M, int Ncols, int Kc) {
  (void)M;
  __shared__ _Float16 As[128 * LDT];   // [row][k]  row-major
  __shared__ _Float16 Bs[128 * LDT];   // [col][k]  transposed so B-fragments are contiguous

  const int tid  = threadIdx.x;
  const int lane = tid & 31;
  const int wave = tid >> 5;
  const int wm = wave >> 2, wn = wave & 3;
  const int m = lane & 15, half = lane >> 4;
  const int row0 = blockIdx.y * 128;
  const int col0 = blockIdx.x * 128;

  v8f acc[4][2];
#pragma unroll
  for (int i = 0; i < 4; ++i)
#pragma unroll
    for (int j = 0; j < 2; ++j)
#pragma unroll
      for (int r = 0; r < 8; ++r) acc[i][j][r] = 0.0f;

  const int ar = tid >> 1, ac = (tid & 1) * 16;   // A: 16 elems / thread
  const int br = tid >> 3, bc = (tid & 7) * 16;   // B: 16 elems / thread

  for (int k0 = 0; k0 < Kc; k0 += 32) {
    // ---- stage A tile 128x32 -> LDS (convert to f16) ----
    {
      const TA* src = A + (size_t)(row0 + ar) * Kc + k0 + ac;
      _Float16 t16[16];
      if constexpr (sizeof(TA) == 4) {
        f4v a0 = *(const f4v*)(src + 0);
        f4v a1 = *(const f4v*)(src + 4);
        f4v a2 = *(const f4v*)(src + 8);
        f4v a3 = *(const f4v*)(src + 12);
#pragma unroll
        for (int i = 0; i < 4; ++i) {
          t16[i]      = (_Float16)a0[i];
          t16[4 + i]  = (_Float16)a1[i];
          t16[8 + i]  = (_Float16)a2[i];
          t16[12 + i] = (_Float16)a3[i];
        }
      } else {
        h8v a0 = *(const h8v*)(src + 0);
        h8v a1 = *(const h8v*)(src + 8);
#pragma unroll
        for (int i = 0; i < 8; ++i) { t16[i] = a0[i]; t16[8 + i] = a1[i]; }
      }
      h8v lo, hi;
#pragma unroll
      for (int i = 0; i < 8; ++i) { lo[i] = t16[i]; hi[i] = t16[8 + i]; }
      h8v* dst = (h8v*)&As[ar * LDT + ac];
      dst[0] = lo; dst[1] = hi;
    }
    // ---- stage B tile 32x128 -> LDS transposed Bs[col][k] ----
    {
      const float* src = Bw + (size_t)(k0 + br) * Ncols + col0 + bc;
      f4v b0 = *(const f4v*)(src + 0);
      f4v b1 = *(const f4v*)(src + 4);
      f4v b2 = *(const f4v*)(src + 8);
      f4v b3 = *(const f4v*)(src + 12);
      float tf[16];
#pragma unroll
      for (int i = 0; i < 4; ++i) {
        tf[i] = b0[i]; tf[4 + i] = b1[i]; tf[8 + i] = b2[i]; tf[12 + i] = b3[i];
      }
#pragma unroll
      for (int i = 0; i < 16; ++i) Bs[(bc + i) * LDT + br] = (_Float16)tf[i];
    }
    if (k0 + 32 < Kc) {   // hint next tiles toward caches (global_prefetch_b8)
      __builtin_prefetch(A + (size_t)(row0 + ar) * Kc + k0 + 32 + ac, 0, 1);
      __builtin_prefetch(Bw + (size_t)(k0 + 32 + br) * Ncols + col0 + bc, 0, 1);
    }
    __syncthreads();

    // ---- fragments + 8 WMMAs ----
    v16h afr[4], bfr[2];
#pragma unroll
    for (int sm = 0; sm < 4; ++sm) {
      const h8v* p = (const h8v*)&As[(wm * 64 + sm * 16 + m) * LDT + half * 16];
      const h8v* q = (const h8v*)&As[(wm * 64 + sm * 16 + m) * LDT + half * 16 + 8];
      afr[sm] = combine16(*p, *q);
    }
#pragma unroll
    for (int sn = 0; sn < 2; ++sn) {
      const h8v* p = (const h8v*)&Bs[(wn * 32 + sn * 16 + m) * LDT + half * 16];
      bfr[sn] = combine16(p[0], p[1]);
    }
#pragma unroll
    for (int sm = 0; sm < 4; ++sm)
#pragma unroll
      for (int sn = 0; sn < 2; ++sn)
        acc[sm][sn] = wmma_f16(afr[sm], bfr[sn], acc[sm][sn]);
    __syncthreads();
  }

  // ---- write C (C/D layout: row = half*8 + r, col = lane&15) ----
#pragma unroll
  for (int sm = 0; sm < 4; ++sm)
#pragma unroll
    for (int sn = 0; sn < 2; ++sn)
#pragma unroll
      for (int r = 0; r < 8; ++r) {
        int gr = row0 + wm * 64 + sm * 16 + half * 8 + r;
        int gc = col0 + wn * 32 + sn * 16 + m;
        C[(size_t)gr * Ncols + gc] = (TO)acc[sm][sn][r];
      }
}

// =====================================================================
// positions_from_segment_ids
// =====================================================================
__global__ void pos_kernel(const int* __restrict__ seg, int* __restrict__ pos, int T) {
  int b = blockIdx.x;
  __shared__ int offS;
  if (threadIdx.x == 0) {
    int best = seg[(size_t)b * T]; int bi = 0;
    for (int t = 1; t < T; ++t) {
      int v = seg[(size_t)b * T + t];
      if (v > best) { best = v; bi = t; }
    }
    offS = bi;
  }
  __syncthreads();
  for (int t = threadIdx.x; t < T; t += blockDim.x) {
    int sv = seg[(size_t)b * T + t];
    pos[(size_t)b * T + t] = (sv != 0) ? (t - offS) : (1 << 30);
  }
}

// =====================================================================
// RMS-norm + RoPE + transpose [B,T,NH,H] -> [B,NH,T,H] (f16 in/out).
// One wave per row of H=128; lane holds elements {l, l+32, l+64, l+96}
// so RoPE pairs (i, i+64) stay within a lane.
// =====================================================================
__global__ __launch_bounds__(128) void norm_rope_transpose(
    const _Float16* __restrict__ src, _Float16* __restrict__ dst,
    const float* __restrict__ scale, const int* __restrict__ pos,
    int T, int NH, int doNormRope) {
  int gw = blockIdx.x * 4 + (threadIdx.x >> 5);
  int lane = threadIdx.x & 31;
  int h  = gw % NH;
  int bt = gw / NH;
  int t = bt % T;
  int b = bt / T;
  const _Float16* s = src + ((size_t)bt * NH + h) * 128;
  float x0 = (float)s[lane];
  float x1 = (float)s[lane + 32];
  float x2 = (float)s[lane + 64];
  float x3 = (float)s[lane + 96];
  if (doNormRope) {
    float ss = x0 * x0 + x1 * x1 + x2 * x2 + x3 * x3;
    ss += __shfl_xor(ss, 1);
    ss += __shfl_xor(ss, 2);
    ss += __shfl_xor(ss, 4);
    ss += __shfl_xor(ss, 8);
    ss += __shfl_xor(ss, 16);
    float rinv = rsqrtf(ss * (1.0f / 128.0f) + 1e-6f);
    x0 *= scale[lane] * rinv;
    x1 *= scale[lane + 32] * rinv;
    x2 *= scale[lane + 64] * rinv;
    x3 *= scale[lane + 96] * rinv;
    float p  = (float)pos[(size_t)b * T + t];
    float a0 = p * __powf(1.0e6f, -(float)lane * (1.0f / 64.0f));
    float a1 = p * __powf(1.0e6f, -(float)(lane + 32) * (1.0f / 64.0f));
    float s0 = __sinf(a0), c0 = __cosf(a0);
    float s1 = __sinf(a1), c1 = __cosf(a1);
    float o0 = x0 * c0 - x2 * s0;
    float o2 = x2 * c0 + x0 * s0;
    float o1 = x1 * c1 - x3 * s1;
    float o3 = x3 * c1 + x1 * s1;
    x0 = o0; x1 = o1; x2 = o2; x3 = o3;
  }
  _Float16* d = dst + ((size_t)(b * NH + h) * T + t) * 128;
  d[lane]      = (_Float16)x0;
  d[lane + 32] = (_Float16)x1;
  d[lane + 64] = (_Float16)x2;
  d[lane + 96] = (_Float16)x3;
}

// =====================================================================
// Flash attention (online softmax), GQA. Block = 4 waves, 64 queries.
// Each wave owns 16 queries; key tiles of 32; WMMA f16 for QK^T and PV.
// K tiles are double-buffered via CDNA5 async global->LDS copies
// (GLOBAL_LOAD_ASYNC_TO_LDS_B128, ASYNCcnt-tracked): the next tile's
// copy is in flight while the current tile is consumed by WMMA.
// =====================================================================
__global__ __launch_bounds__(128) void flash_attn(
    const _Float16* __restrict__ Qh, const _Float16* __restrict__ Kh,
    const _Float16* __restrict__ Vh, const int* __restrict__ pos,
    const int* __restrict__ seg, _Float16* __restrict__ Oout,
    int T, int Nh, int Kvh) {
  __shared__ _Float16 Ks[2][32 * LDK];    // [buf][key][hdim], async-filled, double-buffered
  __shared__ _Float16 Vs[128 * LDT];      // [hdim][key]  (transposed)
  __shared__ _Float16 Ps[4][16 * LDT];    // per-wave P scratch [qrow][key]
  __shared__ int poskS[32];
  __shared__ int segkS[32];

  const int tid = threadIdx.x, lane = tid & 31, wave = tid >> 5;
  const int m = lane & 15, half = lane >> 4;
  const int b = blockIdx.z, n = blockIdx.y;
  const int kvh = n / (Nh / Kvh);
  const int qblk = blockIdx.x * 64;
  const int q0 = qblk + wave * 16;

  const _Float16* Qb = Qh + (size_t)(b * Nh + n) * T * 128;
  const _Float16* Kb = Kh + (size_t)(b * Kvh + kvh) * T * 128;
  const _Float16* Vb = Vh + (size_t)(b * Kvh + kvh) * T * 128;

  // Q fragments (A-layout) resident in VGPRs
  v16h qf[4];
#pragma unroll
  for (int kb = 0; kb < 4; ++kb) {
    const h8v* p = (const h8v*)(Qb + (size_t)(q0 + m) * 128 + kb * 32 + half * 16);
    const h8v* q = (const h8v*)(Qb + (size_t)(q0 + m) * 128 + kb * 32 + half * 16 + 8);
    qf[kb] = combine16(*p, *q);
  }

  float mrow[8], lrow[8];
  int pq[8], sq[8];
  v8f Oacc[8];
#pragma unroll
  for (int r = 0; r < 8; ++r) {
    int t = q0 + half * 8 + r;
    pq[r] = pos[(size_t)b * T + t];
    sq[r] = seg[(size_t)b * T + t];
    mrow[r] = -3.0e38f; lrow[r] = 0.0f;
  }
#pragma unroll
  for (int nt = 0; nt < 8; ++nt)
#pragma unroll
    for (int r = 0; r < 8; ++r) Oacc[nt][r] = 0.0f;

  const float scaleH = 0.08838834764831845f;  // 128^-0.5
  const int send = qblk + 64;                 // causal key limit for this block

  // staging coordinates (shared by async K copies and V transpose)
  const int sRow = tid >> 2;            // 0..31 key row
  const int hCol = (tid & 3) * 32;      // 0/32/64/96 head-dim chunk

  // ---- prologue: async-stage K tile 0 into Ks[0] ----
  {
    const _Float16* ksrc = Kb + (size_t)sRow * 128 + hCol;
#pragma unroll
    for (int i = 0; i < 4; ++i)
      async_load_b128((uint32_t)(uintptr_t)&Ks[0][sRow * LDK + hCol + i * 8],
                      ksrc + i * 8);
  }

  for (int s0 = 0; s0 < send; s0 += 32) {
    const int buf = (s0 >> 5) & 1;
    // ---- cooperative stage: V tile (transposed) + pos/seg ----
    {
      const h8v* vsrc = (const h8v*)(Vb + (size_t)(s0 + sRow) * 128 + hCol);
      h8v v0 = vsrc[0], v1 = vsrc[1], v2 = vsrc[2], v3 = vsrc[3];
#pragma unroll
      for (int i = 0; i < 8; ++i) {
        Vs[(hCol + i) * LDT + sRow]      = v0[i];
        Vs[(hCol + 8 + i) * LDT + sRow]  = v1[i];
        Vs[(hCol + 16 + i) * LDT + sRow] = v2[i];
        Vs[(hCol + 24 + i) * LDT + sRow] = v3[i];
      }
      if (tid < 32) {
        poskS[tid] = pos[(size_t)b * T + s0 + tid];
        segkS[tid] = seg[(size_t)b * T + s0 + tid];
      }
    }
    // ---- kick off async copy of the NEXT K tile, then wait for current ----
    if (s0 + 32 < send) {
      const _Float16* ksrc = Kb + (size_t)(s0 + 32 + sRow) * 128 + hCol;
#pragma unroll
      for (int i = 0; i < 4; ++i)
        async_load_b128((uint32_t)(uintptr_t)&Ks[buf ^ 1][sRow * LDK + hCol + i * 8],
                        ksrc + i * 8);
      // 4 copies for the next tile may remain in flight; current tile's 4 are
      // older and async loads complete in order -> asynccnt<=4 means current done.
      asm volatile("s_wait_asynccnt 0x4" ::: "memory");
    } else {
      asm volatile("s_wait_asynccnt 0x0" ::: "memory");
    }
    __syncthreads();

    // ---- S = Q K^T  (2 key-subtiles x 4 K-depth wmma) ----
    v8f sacc[2];
#pragma unroll
    for (int sn = 0; sn < 2; ++sn)
#pragma unroll
      for (int r = 0; r < 8; ++r) sacc[sn][r] = 0.0f;
#pragma unroll
    for (int sn = 0; sn < 2; ++sn)
#pragma unroll
      for (int kb = 0; kb < 4; ++kb) {
        const h8v* p = (const h8v*)&Ks[buf][(sn * 16 + m) * LDK + kb * 32 + half * 16];
        v16h bf = combine16(p[0], p[1]);
        sacc[sn] = wmma_f16(qf[kb], bf, sacc[sn]);
      }

    // ---- mask + online softmax (row lives in a 16-lane half) ----
    int pk0 = poskS[m],      pk1 = poskS[16 + m];
    int sk0 = segkS[m],      sk1 = segkS[16 + m];
#pragma unroll
    for (int r = 0; r < 8; ++r) {
      float v0 = sacc[0][r] * scaleH;
      float v1 = sacc[1][r] * scaleH;
      v0 = (pk0 <= pq[r] && sk0 == sq[r]) ? v0 : -3.0e38f;
      v1 = (pk1 <= pq[r] && sk1 == sq[r]) ? v1 : -3.0e38f;
      float tmax = fmaxf(v0, v1);
      tmax = fmaxf(tmax, __shfl_xor(tmax, 1));
      tmax = fmaxf(tmax, __shfl_xor(tmax, 2));
      tmax = fmaxf(tmax, __shfl_xor(tmax, 4));
      tmax = fmaxf(tmax, __shfl_xor(tmax, 8));
      float mn = fmaxf(mrow[r], tmax);
      float alpha = __expf(mrow[r] - mn);
      float p0 = __expf(v0 - mn);
      float p1 = __expf(v1 - mn);
      float rs = p0 + p1;
      rs += __shfl_xor(rs, 1);
      rs += __shfl_xor(rs, 2);
      rs += __shfl_xor(rs, 4);
      rs += __shfl_xor(rs, 8);
      lrow[r] = lrow[r] * alpha + rs;
      mrow[r] = mn;
      Ps[wave][(half * 8 + r) * LDT + m]      = (_Float16)p0;
      Ps[wave][(half * 8 + r) * LDT + 16 + m] = (_Float16)p1;
#pragma unroll
      for (int nt = 0; nt < 8; ++nt) Oacc[nt][r] *= alpha;
    }

    // P redistribution: C-layout in LDS -> A-fragment (in-wave RAW: DS in-order + wait)
    asm volatile("s_wait_dscnt 0x0" ::: "memory");
    const h8v* pp0 = (const h8v*)&Ps[wave][m * LDT + half * 16];
    const h8v* pp1 = (const h8v*)&Ps[wave][m * LDT + half * 16 + 8];
    v16h pf = combine16(*pp0, *pp1);

    // ---- O += P * V  (8 head-dim subtiles) ----
#pragma unroll
    for (int nt = 0; nt < 8; ++nt) {
      const h8v* vp = (const h8v*)&Vs[(nt * 16 + m) * LDT + half * 16];
      v16h vf = combine16(vp[0], vp[1]);
      Oacc[nt] = wmma_f16(pf, vf, Oacc[nt]);
    }
    __syncthreads();
  }

  // ---- epilogue: O/l -> [B,T,N,H] f16 ----
#pragma unroll
  for (int r = 0; r < 8; ++r) {
    float inv = 1.0f / lrow[r];
    int t = q0 + half * 8 + r;
    _Float16* dst = Oout + ((size_t)(b * T + t) * Nh + n) * 128;
#pragma unroll
    for (int nt = 0; nt < 8; ++nt) dst[nt * 16 + m] = (_Float16)(Oacc[nt][r] * inv);
  }
}

// =====================================================================
// launch
// =====================================================================
extern "C" void kernel_launch(void* const* d_in, const int* in_sizes, int n_in,
                              void* d_out, int out_size, void* d_ws, size_t ws_size,
                              hipStream_t stream) {
  (void)in_sizes; (void)n_in; (void)out_size; (void)ws_size;
  constexpr int Bb = 2, T = 2048, D = 1024, Nh = 16, Kvh = 8, H = 128;
  constexpr int BT = Bb * T;

  const float* x   = (const float*)d_in[0];
  const int*   seg = (const int*)d_in[1];
  const float* wq  = (const float*)d_in[2];
  const float* wk  = (const float*)d_in[3];
  const float* wv  = (const float*)d_in[4];
  const float* wo  = (const float*)d_in[5];
  const float* qsc = (const float*)d_in[6];
  const float* ksc = (const float*)d_in[7];
  float* out = (float*)d_out;

  uintptr_t p = (uintptr_t)d_ws;
  auto take = [&](size_t bytes) -> uintptr_t {
    uintptr_t cur = (p + 255) & ~(uintptr_t)255;
    p = cur + bytes;
    return cur;
  };
  int*      pos  = (int*)take((size_t)BT * 4);
  _Float16* qraw = (_Float16*)take((size_t)BT * Nh * H * 2);   // [B,T,N,H]
  _Float16* kraw = (_Float16*)take((size_t)BT * Kvh * H * 2);  // [B,T,K,H]
  _Float16* vraw = (_Float16*)take((size_t)BT * Kvh * H * 2);
  _Float16* qT   = (_Float16*)take((size_t)BT * Nh * H * 2);   // [B,N,T,H]
  _Float16* kT   = (_Float16*)take((size_t)BT * Kvh * H * 2);  // [B,K,T,H]
  _Float16* vT   = (_Float16*)take((size_t)BT * Kvh * H * 2);
  _Float16* attn = (_Float16*)take((size_t)BT * Nh * H * 2);   // [B,T,N,H]

  pos_kernel<<<Bb, 256, 0, stream>>>(seg, pos, T);

  // QKV projections (f32 in, f32-accumulate via f16 wmma, f16 out)
  gemm_wmma<float, _Float16><<<dim3((Nh * H) / 128, BT / 128), 256, 0, stream>>>(
      x, wq, qraw, BT, Nh * H, D);
  gemm_wmma<float, _Float16><<<dim3((Kvh * H) / 128, BT / 128), 256, 0, stream>>>(
      x, wk, kraw, BT, Kvh * H, D);
  gemm_wmma<float, _Float16><<<dim3((Kvh * H) / 128, BT / 128), 256, 0, stream>>>(
      x, wv, vraw, BT, Kvh * H, D);

  // RMS-norm + RoPE + head-major transpose
  norm_rope_transpose<<<(BT * Nh) / 4, 128, 0, stream>>>(qraw, qT, qsc, pos, T, Nh, 1);
  norm_rope_transpose<<<(BT * Kvh) / 4, 128, 0, stream>>>(kraw, kT, ksc, pos, T, Kvh, 1);
  norm_rope_transpose<<<(BT * Kvh) / 4, 128, 0, stream>>>(vraw, vT, ksc, pos, T, Kvh, 0);

  // Flash attention (async double-buffered K tiles)
  flash_attn<<<dim3(T / 64, Nh, Bb), 128, 0, stream>>>(qT, kT, vT, pos, seg, attn, T, Nh, Kvh);

  // Output projection (f16 activations, f32 out)
  gemm_wmma<_Float16, float><<<dim3(D / 128, BT / 128), 256, 0, stream>>>(
      attn, wo, out, BT, D, Nh * H);
}
